// MultiplexInductiveSmoother_57303453663686
// MI455X (gfx1250) — matrix operand order repaired and checked by reading, not compiled, not run on hardware
//
#include <hip/hip_runtime.h>

// ---------------------------------------------------------------------------
// MultiplexInductiveSmoother for MI455X (gfx1250, wave32, WMMA bf16)
//
// Sizes: P=512, D=128, NF=NR=50000, ND=100000, EF=ER=400000, out = 1152 f32.
// Workspace requirement: ~110 MB (node feats stored bf16).
//
// Key structure:
//  - node MLPs (the only FLOP-heavy part, ~105 GFLOP) run on
//    v_wmma_f32_16x16x32_bf16 with pre-transposed bf16 weights so B fragments
//    are direct global_load_b128 (L2-resident), A fragments ds_load_b128.
//  - edge attention is algebraically collapsed: logit_e = d.kq + ckb,
//    v_form = (sum c_e d_e) @ v_w + (sum c_e) v_b  -> gathers + dots only.
//  - delta = z - attn@feats (softmax sums to 1), feats kept bf16.
// ---------------------------------------------------------------------------

#define NF_N 50000
#define EF_N 400000
#define BASELINE_C 6.0f

typedef __attribute__((ext_vector_type(16))) __bf16 v16bf;
typedef __attribute__((ext_vector_type(8)))  float  v8f;

union BF16Frag { v16bf v; unsigned u[8]; };

__device__ __forceinline__ unsigned short f2bf(float f) {
  unsigned u = __float_as_uint(f);
  unsigned r = u + 0x7FFFu + ((u >> 16) & 1u);   // round-to-nearest-even
  return (unsigned short)(r >> 16);
}
__device__ __forceinline__ float bf2f(unsigned short h) {
  return __uint_as_float(((unsigned)h) << 16);
}
__device__ __forceinline__ unsigned fenc(float f) {  // order-preserving f32->u32
  unsigned u = __float_as_uint(f);
  return (u & 0x80000000u) ? ~u : (u | 0x80000000u);
}
__device__ __forceinline__ float fdec(unsigned e) {
  unsigned u = (e & 0x80000000u) ? (e & 0x7fffffffu) : ~e;
  return __uint_as_float(u);
}
__device__ __forceinline__ v8f v8f_zero() {
  v8f z = {0.f, 0.f, 0.f, 0.f, 0.f, 0.f, 0.f, 0.f};
  return z;
}

// ------------------------------- utility -----------------------------------
__global__ void mis_zero(float* __restrict__ p, long n) {
  for (long i = blockIdx.x * (long)blockDim.x + threadIdx.x; i < n;
       i += (long)gridDim.x * blockDim.x) p[i] = 0.f;
}

// dst[n*K + k] = bf16(src[k*N + n]); one 32x32 tile per block (256 thr)
__global__ __launch_bounds__(256) void mis_cvt_bf16_t(
    const float* __restrict__ src, unsigned short* __restrict__ dst, int K, int N)
{
  __shared__ float tile[32][33];
  int tilesx = N >> 5;
  int tx = blockIdx.x % tilesx;
  int ty = blockIdx.x / tilesx;
  int n0 = tx << 5, k0 = ty << 5;
  int c = threadIdx.x & 31, r8 = threadIdx.x >> 5;
  for (int r = r8; r < 32; r += 8)
    tile[r][c] = src[(long)(k0 + r) * N + n0 + c];
  __syncthreads();
  for (int r = r8; r < 32; r += 8)
    dst[(long)(n0 + r) * K + k0 + c] = f2bf(tile[c][r]);
}

// --------------------- z, q, kq, ckb, c1 (one block) ------------------------
__global__ __launch_bounds__(512) void mis_compute_z(
    const float* __restrict__ tf,
    const float* __restrict__ fw1, const float* __restrict__ fb1, const float* __restrict__ fa1,
    const float* __restrict__ fw2, const float* __restrict__ fb2,
    const float* __restrict__ rw1, const float* __restrict__ rb1, const float* __restrict__ ra1,
    const float* __restrict__ rw2, const float* __restrict__ rb2,
    const float* __restrict__ q_w, const float* __restrict__ q_b,
    const float* __restrict__ k_w, const float* __restrict__ k_b,
    const float* __restrict__ lemb, const float* __restrict__ aw1, const float* __restrict__ ab1,
    float* __restrict__ zg, float* __restrict__ kqg, float* __restrict__ ckbg,
    float* __restrict__ c1f, float* __restrict__ c1r)
{
  __shared__ float sx[512], sh[512], sz[512], sq[128];
  int t = threadIdx.x;
  sx[t] = tf[t];
  __syncthreads();
  float a = fa1[0];
  float s = fb1[t];
  for (int i = 0; i < 512; ++i) s += sx[i] * fw1[i * 512 + t];
  sh[t] = s >= 0.f ? s : a * s;
  __syncthreads();
  float zf = fb2[t];
  for (int i = 0; i < 512; ++i) zf += sh[i] * fw2[i * 512 + t];
  __syncthreads();
  a = ra1[0];
  s = rb1[t];
  for (int i = 0; i < 512; ++i) s += sx[i] * rw1[i * 512 + t];
  sh[t] = s >= 0.f ? s : a * s;
  __syncthreads();
  float zr = rb2[t];
  for (int i = 0; i < 512; ++i) zr += sh[i] * rw2[i * 512 + t];
  float z = 0.5f * (zf + zr);
  sz[t] = z;
  zg[t] = z;
  __syncthreads();
  if (t < 128) {
    float qv = q_b[t];
    for (int i = 0; i < 512; ++i) qv += sz[i] * q_w[i * 128 + t];
    sq[t] = qv;
  }
  __syncthreads();
  const float rs = 0.08838834764831845f;  // 1/sqrt(128)
  if (t < 128) {
    float kv = 0.f;
    for (int j = 0; j < 128; ++j) kv += k_w[t * 128 + j] * sq[j];
    kqg[t] = kv * rs;
  }
  if (t == 0) {
    float cb = 0.f;
    for (int j = 0; j < 128; ++j) cb += k_b[j] * sq[j];
    *ckbg = cb * rs;
  }
  if (t < 256) {
    int layer = t >> 7, j = t & 127;
    float c = ab1[j];
    for (int i = 0; i < 512; ++i) c += sz[i] * aw1[i * 128 + j];
    for (int i = 0; i < 16; ++i) c += lemb[layer * 16 + i] * aw1[(1024 + i) * 128 + j];
    if (layer) c1r[j] = c; else c1f[j] = c;
  }
}

// ----------------- fused node MLP + attn-logit head (WMMA) ------------------
// 32 rows/block, 256 threads (8 waves). Weights are pre-transposed bf16:
// Wt[n][k], so a B fragment = 2x global_load_b128 per lane (L2 hits).
// LDS arena (58624 B):
//   GEMM phase   : s_a  = 32 x 520 bf16 (33280 B) at 0
//   logits phase : s_h  (32x132 f32, 16896) at 0      | overlays s_a
//                  s_h2 (32x68  f32,  8704) at 16896
//                  s_w2 (128x64 f32, 32768) at 25600
//                  s_w3 (64 f32,      256)  at 58368
__global__ __launch_bounds__(256) void mis_node_mlp(
    const float* __restrict__ X, int N,
    const unsigned short* __restrict__ Wt1, const float* __restrict__ b1, const float* __restrict__ a1p,
    const unsigned short* __restrict__ Wt2, const float* __restrict__ b2,
    const unsigned short* __restrict__ W1bt, const float* __restrict__ c1,
    const float* __restrict__ aw2, const float* __restrict__ ab2,
    const float* __restrict__ aw3, const float* __restrict__ ab3,
    const float* __restrict__ diffw,
    unsigned short* __restrict__ featsOut, float* __restrict__ nlogOut)
{
  __shared__ __align__(16) char smem[58624];
  unsigned short* s_a  = (unsigned short*)smem;       // 32 x 520
  float*          s_h  = (float*)smem;                // 32 x 132
  float*          s_h2 = (float*)(smem + 16896);      // 32 x 68
  float*          s_w2 = (float*)(smem + 25600);      // 128 x 64
  float*          s_w3 = (float*)(smem + 58368);      // 64

  const int tid  = threadIdx.x;
  const int lane = tid & 31;
  const int w    = tid >> 5;
  const int rowBase = blockIdx.x * 32;
  const float a1v = a1p[0];
  const int mtile = w & 1;
  const int ntb   = (w >> 1) << 3;     // 8 n-tiles per wave
  const int lhi8  = (lane >> 4) << 3;  // 0 / 8
  const int lhi16 = (lane >> 4) << 4;  // 0 / 16
  const int l15   = lane & 15;

  // Stage A: 32 rows x 512, fp32 -> bf16, vectorized (float4 -> uint2)
  for (int i = tid; i < 32 * 128; i += 256) {
    int r = i >> 7, c = (i & 127) << 2;
    int gr = rowBase + r;
    float4 v = make_float4(0.f, 0.f, 0.f, 0.f);
    if (gr < N) v = *(const float4*)(X + (long)gr * 512 + c);
    unsigned p0 = ((unsigned)f2bf(v.y) << 16) | f2bf(v.x);
    unsigned p1 = ((unsigned)f2bf(v.w) << 16) | f2bf(v.z);
    *(uint2*)(s_a + r * 520 + c) = make_uint2(p0, p1);
  }
  __syncthreads();

  for (int pass = 0; pass < 2; ++pass) {
    const unsigned short* Wt = pass ? Wt2 : Wt1;
    v8f acc[8];
#pragma unroll
    for (int i = 0; i < 8; ++i) acc[i] = v8f_zero();
    for (int kb = 0; kb < 16; ++kb) {
      // A fragment (16x32 bf16, ISA layout): two aligned b128 LDS loads
      BF16Frag af;
      {
        const unsigned short* pa = s_a + (mtile * 16 + l15) * 520 + kb * 32 + lhi8;
        uint4 q0 = *(const uint4*)pa;
        uint4 q1 = *(const uint4*)(pa + 16);
        af.u[0] = q0.x; af.u[1] = q0.y; af.u[2] = q0.z; af.u[3] = q0.w;
        af.u[4] = q1.x; af.u[5] = q1.y; af.u[6] = q1.z; af.u[7] = q1.w;
      }
#pragma unroll
      for (int nt = 0; nt < 8; ++nt) {
        int n = (ntb + nt) * 16 + l15;
        const unsigned short* pb = Wt + (long)n * 512 + kb * 32 + lhi8;
        uint4 q0 = *(const uint4*)pb;        // K = base+0..7
        uint4 q1 = *(const uint4*)(pb + 16); // K = base+16..23
        BF16Frag bfr;
        bfr.u[0] = q0.x; bfr.u[1] = q0.y; bfr.u[2] = q0.z; bfr.u[3] = q0.w;
        bfr.u[4] = q1.x; bfr.u[5] = q1.y; bfr.u[6] = q1.z; bfr.u[7] = q1.w;
        acc[nt] = __builtin_amdgcn_wmma_f32_16x16x32_bf16(
            false, af.v, false, bfr.v, (short)0, acc[nt], false, false);
      }
    }
    __syncthreads();  // everyone done reading s_a for this pass
    // Epilogue: bias (+PReLU pass0), write back to s_a as bf16.
#pragma unroll
    for (int nt = 0; nt < 8; ++nt) {
      int col = (ntb + nt) * 16 + l15;
      float bv = pass ? b2[col] : b1[col];
#pragma unroll
      for (int r = 0; r < 8; ++r) {
        int row = mtile * 16 + r + lhi8;
        float v = acc[nt][r] + bv;
        if (pass == 0) {
          v = v >= 0.f ? v : a1v * v;
        } else {
          int gr = rowBase + row;
          if (gr < N) featsOut[(long)gr * 512 + col] = f2bf(v);
        }
        s_a[row * 520 + col] = f2bf(v);
      }
    }
    __syncthreads();
  }

  // GEMM3: logits head, feats(32x512) @ W1bt(128x512 transposed).
  // Wave w owns ntile = w, both m-tiles.
  v8f acc3[2];
  acc3[0] = v8f_zero();
  acc3[1] = v8f_zero();
  for (int kb = 0; kb < 16; ++kb) {
    BF16Frag bfr;
    {
      int n = w * 16 + l15;
      const unsigned short* pb = W1bt + (long)n * 512 + kb * 32 + lhi8;
      uint4 q0 = *(const uint4*)pb;
      uint4 q1 = *(const uint4*)(pb + 16);
      bfr.u[0] = q0.x; bfr.u[1] = q0.y; bfr.u[2] = q0.z; bfr.u[3] = q0.w;
      bfr.u[4] = q1.x; bfr.u[5] = q1.y; bfr.u[6] = q1.z; bfr.u[7] = q1.w;
    }
#pragma unroll
    for (int m = 0; m < 2; ++m) {
      BF16Frag af;
      const unsigned short* pa = s_a + (m * 16 + l15) * 520 + kb * 32 + lhi8;
      uint4 q0 = *(const uint4*)pa;
      uint4 q1 = *(const uint4*)(pa + 16);
      af.u[0] = q0.x; af.u[1] = q0.y; af.u[2] = q0.z; af.u[3] = q0.w;
      af.u[4] = q1.x; af.u[5] = q1.y; af.u[6] = q1.z; af.u[7] = q1.w;
      acc3[m] = __builtin_amdgcn_wmma_f32_16x16x32_bf16(
          false, af.v, false, bfr.v, (short)0, acc3[m], false, false);
    }
  }
  __syncthreads();  // done with s_a; overlay region becomes live
  // h1 = leaky(feats @ W1b + c1, 0.2)   (c1 folds z-part, lemb-part, b1)
  {
    int col = w * 16 + l15;
    float cv = c1[col];
#pragma unroll
    for (int m = 0; m < 2; ++m) {
#pragma unroll
      for (int r = 0; r < 8; ++r) {
        int row = m * 16 + r + lhi8;
        float v = acc3[m][r] + cv;
        s_h[row * 132 + col] = v >= 0.f ? v : 0.2f * v;
      }
    }
  }
  for (int i = tid; i < 2048; i += 256)
    ((float4*)s_w2)[i] = ((const float4*)aw2)[i];
  if (tid < 16) ((float4*)s_w3)[tid] = ((const float4*)aw3)[tid];
  __syncthreads();
  for (int j = tid; j < 32 * 64; j += 256) {
    int row = j >> 6, col = j & 63;
    float s = ab2[col];
    for (int i = 0; i < 128; ++i) s += s_h[row * 132 + i] * s_w2[i * 64 + col];
    s_h2[row * 68 + col] = s >= 0.f ? s : 0.2f * s;
  }
  __syncthreads();
  if (tid < 32) {
    int row = tid;
    float s = ab3[0];
    for (int i = 0; i < 64; ++i) s += s_h2[row * 68 + i] * s_w3[i];
    int gr = rowBase + row;
    if (gr < N) nlogOut[gr] = s + logf(fmaxf(diffw[gr], 1e-12f));
  }
}

// ------------------------- global softmax stats -----------------------------
__global__ __launch_bounds__(256) void mis_reduce_max_enc(
    const float* __restrict__ x, int n, unsigned* __restrict__ out)
{
  __shared__ unsigned sm[256];
  unsigned m = 0u;
  for (long i = blockIdx.x * (long)blockDim.x + threadIdx.x; i < n;
       i += (long)gridDim.x * blockDim.x) {
    unsigned e = fenc(x[i]);
    m = e > m ? e : m;
  }
  sm[threadIdx.x] = m;
  __syncthreads();
  for (int o = 128; o > 0; o >>= 1) {
    if ((int)threadIdx.x < o && sm[threadIdx.x + o] > sm[threadIdx.x])
      sm[threadIdx.x] = sm[threadIdx.x + o];
    __syncthreads();
  }
  if (threadIdx.x == 0) atomicMax(out, sm[0]);
}

__global__ __launch_bounds__(256) void mis_reduce_sumexp(
    const float* __restrict__ x, int n, const unsigned* __restrict__ maxe,
    float* __restrict__ out)
{
  __shared__ float sm[256];
  float m = fdec(*maxe);
  float s = 0.f;
  for (long i = blockIdx.x * (long)blockDim.x + threadIdx.x; i < n;
       i += (long)gridDim.x * blockDim.x) s += __expf(x[i] - m);
  sm[threadIdx.x] = s;
  __syncthreads();
  for (int o = 128; o > 0; o >>= 1) {
    if ((int)threadIdx.x < o) sm[threadIdx.x] += sm[threadIdx.x + o];
    __syncthreads();
  }
  if (threadIdx.x == 0) atomicAdd(out, sm[0]);
}

// --------------------------- edge pipeline ----------------------------------
// logit_e = d[dst] . kq + ckb ; per-src running max (encoded atomic max)
__global__ __launch_bounds__(256) void mis_edge_logits(
    const int* __restrict__ src, const int* __restrict__ dst,
    const float* __restrict__ drugs, const float* __restrict__ kq,
    const float* __restrict__ ckbp,
    float* __restrict__ elog, unsigned* __restrict__ mx, int E)
{
  int lane4 = threadIdx.x & 3;
  long quad  = (blockIdx.x * (long)blockDim.x + threadIdx.x) >> 2;
  long nquad = ((long)gridDim.x * blockDim.x) >> 2;
  float ck = *ckbp;
  for (long e = quad; e < E; e += nquad) {
    int d = dst[e];
    const float4* dp = (const float4*)(drugs + (long)d * 128 + lane4 * 32);
    const float4* qp = (const float4*)(kq + lane4 * 32);
    float s = 0.f;
#pragma unroll
    for (int i = 0; i < 8; ++i) {
      float4 a = dp[i], b = qp[i];
      s += a.x * b.x + a.y * b.y + a.z * b.z + a.w * b.w;
    }
    s += __shfl_xor(s, 1, 4);
    s += __shfl_xor(s, 2, 4);
    if (lane4 == 0) {
      float lg = s + ck;
      elog[e] = lg;
      atomicMax(&mx[src[e]], fenc(lg));
    }
  }
}

__global__ __launch_bounds__(256) void mis_edge_expsum(
    const int* __restrict__ src, const float* __restrict__ elog,
    const unsigned* __restrict__ mx, float* __restrict__ den, int E)
{
  for (long e = blockIdx.x * (long)blockDim.x + threadIdx.x; e < E;
       e += (long)gridDim.x * blockDim.x) {
    int s = src[e];
    atomicAdd(&den[s], __expf(elog[e] - fdec(mx[s])));
  }
}

// S[0:128] = sum_e c_e * drugs[dst_e], S[128] = sum_e c_e
// c_e = node_attn[src] * edge_attn_e * (y_e - 6) * w_e
__global__ __launch_bounds__(256) void mis_edge_accum(
    const int* __restrict__ src, const int* __restrict__ dst,
    const float* __restrict__ y, const float* __restrict__ ww,
    const float* __restrict__ elog, const unsigned* __restrict__ mx,
    const float* __restrict__ den, const float* __restrict__ nlog,
    const unsigned* __restrict__ nmaxp, const float* __restrict__ nsump,
    const float* __restrict__ drugs, float* __restrict__ S, int E)
{
  __shared__ float ls[129];
  for (int i = threadIdx.x; i < 129; i += blockDim.x) ls[i] = 0.f;
  __syncthreads();
  int lane4 = threadIdx.x & 3;
  float nm = fdec(*nmaxp), ns = *nsump;
  float acc[32];
#pragma unroll
  for (int i = 0; i < 32; ++i) acc[i] = 0.f;
  float csum = 0.f;
  long quad  = (blockIdx.x * (long)blockDim.x + threadIdx.x) >> 2;
  long nquad = ((long)gridDim.x * blockDim.x) >> 2;
  for (long e = quad; e < E; e += nquad) {
    int sI = src[e];
    float c = 0.f;
    if (lane4 == 0) {
      float ea = __expf(elog[e] - fdec(mx[sI])) / (den[sI] + 1e-12f);
      float na = __expf(nlog[sI] - nm) / ns;
      c = na * ea * ((y[e] - BASELINE_C) * ww[e]);
    }
    c = __shfl(c, 0, 4);
    int d = dst[e];
    const float4* dp = (const float4*)(drugs + (long)d * 128 + lane4 * 32);
#pragma unroll
    for (int i = 0; i < 8; ++i) {
      float4 a = dp[i];
      acc[4 * i] += c * a.x; acc[4 * i + 1] += c * a.y;
      acc[4 * i + 2] += c * a.z; acc[4 * i + 3] += c * a.w;
    }
    if (lane4 == 0) csum += c;
  }
#pragma unroll
  for (int i = 0; i < 32; ++i) atomicAdd(&ls[lane4 * 32 + i], acc[i]);
  atomicAdd(&ls[128], csum);
  __syncthreads();
  for (int i = threadIdx.x; i < 129; i += blockDim.x) atomicAdd(&S[i], ls[i]);
}

// F[0:512] = sum_n node_attn[n] * feats_bf16[n, :]
__global__ __launch_bounds__(256) void mis_node_wsum(
    const unsigned short* __restrict__ feats, const float* __restrict__ nlog,
    const unsigned* __restrict__ nmaxp, const float* __restrict__ nsump,
    float* __restrict__ F, int n)
{
  float nm = fdec(*nmaxp), ns = *nsump;
  int c0 = threadIdx.x * 2;
  float a0 = 0.f, a1 = 0.f;
  for (int r = blockIdx.x; r < n; r += gridDim.x) {
    float a = __expf(nlog[r] - nm) / ns;
    unsigned pair = *(const unsigned*)(feats + (long)r * 512 + c0);
    a0 += a * bf2f((unsigned short)(pair & 0xffffu));
    a1 += a * bf2f((unsigned short)(pair >> 16));
  }
  atomicAdd(&F[c0], a0);
  atomicAdd(&F[c0 + 1], a1);
}

// ------------------------------- finalize -----------------------------------
__global__ __launch_bounds__(512) void mis_finalize(
    const float* __restrict__ Sf, const float* __restrict__ Sr,
    const float* __restrict__ v_w, const float* __restrict__ v_b,
    const float* __restrict__ trust,
    const float* __restrict__ mw1, const float* __restrict__ mb1,
    const float* __restrict__ mxa, const float* __restrict__ mw2,
    const float* __restrict__ mb2,
    const float* __restrict__ zg, const float* __restrict__ Ff,
    const float* __restrict__ Fr,
    const float* __restrict__ iw1, const float* __restrict__ ib1,
    const float* __restrict__ iap, const float* __restrict__ iw2,
    const float* __restrict__ ib2,
    const float* __restrict__ nw, const float* __restrict__ nb,
    const float* __restrict__ dgate, float* __restrict__ out)
{
  __shared__ float vf[128], vr[128], vp[128], h64[64], l2[2], h512[512], red[512];
  int t = threadIdx.x;
  if (t < 128) {
    float s = Sf[128] * v_b[t];
    for (int i = 0; i < 128; ++i) s += Sf[i] * v_w[i * 128 + t];
    vf[t] = s;
  } else if (t < 256) {
    int j = t - 128;
    float s = Sr[128] * v_b[j];
    for (int i = 0; i < 128; ++i) s += Sr[i] * v_w[i * 128 + j];
    vr[j] = s;
  }
  __syncthreads();
  if (t < 64) {
    float s = mb1[t];
    for (int i = 0; i < 128; ++i) s += vf[i] * mw1[i * 64 + t];
    for (int i = 0; i < 128; ++i) s += vr[i] * mw1[(128 + i) * 64 + t];
    for (int i = 0; i < 3; ++i)   s += trust[i] * mw1[(256 + i) * 64 + t];
    float ma = mxa[0];
    h64[t] = s >= 0.f ? s : ma * s;
  }
  __syncthreads();
  if (t < 2) {
    float s = mb2[t];
    for (int i = 0; i < 64; ++i) s += h64[i] * mw2[i * 2 + t];
    l2[t] = s;
  }
  __syncthreads();
  float mm = fmaxf(l2[0], l2[1]);
  float e0 = __expf(l2[0] - mm), e1 = __expf(l2[1] - mm);
  float wf = e0 / (e0 + e1), wr = e1 / (e0 + e1);
  if (t < 128) {
    vp[t] = wf * vf[t] + wr * vr[t];
    out[512 + t] = vp[t];
  }
  float z = zg[t];
  float d = wf * (z - Ff[t]) + wr * (z - Fr[t]);  // softmax sums to 1
  __syncthreads();
  // delta_mean = delta_gate * LN(delta_raw)
  red[t] = d; __syncthreads();
  for (int o = 256; o > 0; o >>= 1) { if (t < o) red[t] += red[t + o]; __syncthreads(); }
  float mean = red[0] * (1.f / 512.f);
  __syncthreads();
  float dm = d - mean;
  red[t] = dm * dm; __syncthreads();
  for (int o = 256; o > 0; o >>= 1) { if (t < o) red[t] += red[t + o]; __syncthreads(); }
  float inv = rsqrtf(red[0] * (1.f / 512.f) + 1e-5f);
  __syncthreads();
  out[640 + t] = dgate[0] * dm * inv;
  // z_refined = LN(z + mlp2(v_prior, int), norm_w, norm_b)
  float s = ib1[t];
  for (int i = 0; i < 128; ++i) s += vp[i] * iw1[(long)i * 512 + t];
  float ia = iap[0];
  h512[t] = s >= 0.f ? s : ia * s;
  __syncthreads();
  float yv = ib2[t];
  for (int i = 0; i < 512; ++i) yv += h512[i] * iw2[(long)i * 512 + t];
  float u = z + yv;
  red[t] = u; __syncthreads();
  for (int o = 256; o > 0; o >>= 1) { if (t < o) red[t] += red[t + o]; __syncthreads(); }
  float mean2 = red[0] * (1.f / 512.f);
  __syncthreads();
  float um = u - mean2;
  red[t] = um * um; __syncthreads();
  for (int o = 256; o > 0; o >>= 1) { if (t < o) red[t] += red[t + o]; __syncthreads(); }
  float inv2 = rsqrtf(red[0] * (1.f / 512.f) + 1e-5f);
  out[t] = um * inv2 * nw[t] + nb[t];
}

// ------------------------------- launcher -----------------------------------
extern "C" void kernel_launch(void* const* d_in, const int* in_sizes, int n_in,
                              void* d_out, int out_size, void* d_ws, size_t ws_size,
                              hipStream_t stream)
{
  (void)in_sizes; (void)n_in; (void)out_size; (void)ws_size;
  const float* tf    = (const float*)d_in[0];
  const float* formX = (const float*)d_in[1];
  const float* roleX = (const float*)d_in[2];
  const float* f_y   = (const float*)d_in[3];
  const float* f_w   = (const float*)d_in[4];
  const float* r_y   = (const float*)d_in[5];
  const float* r_w   = (const float*)d_in[6];
  const float* fdiff = (const float*)d_in[7];
  const float* rdiff = (const float*)d_in[8];
  const float* trust = (const float*)d_in[9];
  const float* drugs = (const float*)d_in[10];
  const float* fw1 = (const float*)d_in[11]; const float* fb1 = (const float*)d_in[12];
  const float* fa1 = (const float*)d_in[13];
  const float* fw2 = (const float*)d_in[14]; const float* fb2 = (const float*)d_in[15];
  const float* rw1 = (const float*)d_in[16]; const float* rb1 = (const float*)d_in[17];
  const float* ra1 = (const float*)d_in[18];
  const float* rw2 = (const float*)d_in[19]; const float* rb2 = (const float*)d_in[20];
  const float* q_w = (const float*)d_in[21]; const float* q_b = (const float*)d_in[22];
  const float* k_w = (const float*)d_in[23]; const float* k_b = (const float*)d_in[24];
  const float* v_w = (const float*)d_in[25]; const float* v_b = (const float*)d_in[26];
  const float* lemb = (const float*)d_in[27];
  const float* aw1 = (const float*)d_in[28]; const float* ab1 = (const float*)d_in[29];
  const float* aw2 = (const float*)d_in[30]; const float* ab2 = (const float*)d_in[31];
  const float* aw3 = (const float*)d_in[32]; const float* ab3 = (const float*)d_in[33];
  const float* mw1 = (const float*)d_in[34]; const float* mb1 = (const float*)d_in[35];
  const float* mxa = (const float*)d_in[36];
  const float* mw2 = (const float*)d_in[37]; const float* mb2 = (const float*)d_in[38];
  const float* iw1 = (const float*)d_in[39]; const float* ib1 = (const float*)d_in[40];
  const float* iap = (const float*)d_in[41];
  const float* iw2 = (const float*)d_in[42]; const float* ib2 = (const float*)d_in[43];
  const float* nw = (const float*)d_in[44]; const float* nb = (const float*)d_in[45];
  const float* dgate = (const float*)d_in[46];
  const int* f_src = (const int*)d_in[49];
  const int* f_dst = (const int*)d_in[50];
  const int* r_src = (const int*)d_in[51];
  const int* r_dst = (const int*)d_in[52];
  float* out = (float*)d_out;

  const int N = NF_N, E = EF_N;

  // ---- workspace carve (512B aligned) ----
  char* base = (char*)d_ws;
  size_t off = 0;
  auto carve = [&](size_t bytes) -> void* {
    void* p = base + off;
    off = (off + bytes + 511) & ~(size_t)511;
    return p;
  };
  unsigned short* wt1f = (unsigned short*)carve(512 * 512 * 2);  // transposed bf16
  unsigned short* wt2f = (unsigned short*)carve(512 * 512 * 2);
  unsigned short* wt1r = (unsigned short*)carve(512 * 512 * 2);
  unsigned short* wt2r = (unsigned short*)carve(512 * 512 * 2);
  unsigned short* w1bt = (unsigned short*)carve(128 * 512 * 2);  // [n=128][k=512]
  unsigned short* featF = (unsigned short*)carve((size_t)N * 512 * 2);
  unsigned short* featR = (unsigned short*)carve((size_t)N * 512 * 2);
  float* zg  = (float*)carve(512 * 4);
  float* kq  = (float*)carve(128 * 4);
  float* ckb = (float*)carve(16);
  float* c1f = (float*)carve(128 * 4);
  float* c1r = (float*)carve(128 * 4);
  float* nlogF = (float*)carve((size_t)N * 4);
  float* nlogR = (float*)carve((size_t)N * 4);
  float* elogF = (float*)carve((size_t)E * 4);
  float* elogR = (float*)carve((size_t)E * 4);
  size_t stats_off = off;  // everything below is zero-initialized each call
  unsigned* mxF = (unsigned*)carve((size_t)N * 4);
  unsigned* mxR = (unsigned*)carve((size_t)N * 4);
  float* denF = (float*)carve((size_t)N * 4);
  float* denR = (float*)carve((size_t)N * 4);
  unsigned* nmaxF = (unsigned*)carve(16);
  unsigned* nmaxR = (unsigned*)carve(16);
  float* nsumF = (float*)carve(16);
  float* nsumR = (float*)carve(16);
  float* SF = (float*)carve(129 * 4);
  float* SR = (float*)carve(129 * 4);
  float* FF = (float*)carve(512 * 4);
  float* FR = (float*)carve(512 * 4);
  size_t stats_bytes = off - stats_off;

  // ---- 0. zero the accumulator/stat region ----
  {
    long cnt = (long)(stats_bytes / 4);
    int blk = (int)((cnt + 255) / 256);
    if (blk > 2048) blk = 2048;
    mis_zero<<<blk, 256, 0, stream>>>((float*)(base + stats_off), cnt);
  }
  // ---- 1. weights -> bf16, transposed (L2-resident afterwards) ----
  mis_cvt_bf16_t<<<256, 256, 0, stream>>>(fw1, wt1f, 512, 512);
  mis_cvt_bf16_t<<<256, 256, 0, stream>>>(fw2, wt2f, 512, 512);
  mis_cvt_bf16_t<<<256, 256, 0, stream>>>(rw1, wt1r, 512, 512);
  mis_cvt_bf16_t<<<256, 256, 0, stream>>>(rw2, wt2r, 512, 512);
  mis_cvt_bf16_t<<<64, 256, 0, stream>>>(aw1 + 512L * 128, w1bt, 512, 128);
  // ---- 2. z / kq / ckb / c1 ----
  mis_compute_z<<<1, 512, 0, stream>>>(tf, fw1, fb1, fa1, fw2, fb2,
                                       rw1, rb1, ra1, rw2, rb2,
                                       q_w, q_b, k_w, k_b, lemb, aw1, ab1,
                                       zg, kq, ckb, c1f, c1r);
  // ---- 3. node MLPs + logit head (WMMA) ----
  int nmblk = (N + 31) / 32;
  mis_node_mlp<<<nmblk, 256, 0, stream>>>(formX, N, wt1f, fb1, fa1, wt2f, fb2,
                                          w1bt, c1f, aw2, ab2, aw3, ab3, fdiff,
                                          featF, nlogF);
  mis_node_mlp<<<nmblk, 256, 0, stream>>>(roleX, N, wt1r, rb1, ra1, wt2r, rb2,
                                          w1bt, c1r, aw2, ab2, aw3, ab3, rdiff,
                                          featR, nlogR);
  // ---- 4. node softmax stats ----
  mis_reduce_max_enc<<<128, 256, 0, stream>>>(nlogF, N, nmaxF);
  mis_reduce_max_enc<<<128, 256, 0, stream>>>(nlogR, N, nmaxR);
  mis_reduce_sumexp<<<128, 256, 0, stream>>>(nlogF, N, nmaxF, nsumF);
  mis_reduce_sumexp<<<128, 256, 0, stream>>>(nlogR, N, nmaxR, nsumR);
  // ---- 5. edge logits + per-src max ----
  mis_edge_logits<<<4096, 256, 0, stream>>>(f_src, f_dst, drugs, kq, ckb, elogF, mxF, E);
  mis_edge_logits<<<4096, 256, 0, stream>>>(r_src, r_dst, drugs, kq, ckb, elogR, mxR, E);
  // ---- 6. edge softmax denominators ----
  mis_edge_expsum<<<2048, 256, 0, stream>>>(f_src, elogF, mxF, denF, E);
  mis_edge_expsum<<<2048, 256, 0, stream>>>(r_src, elogR, mxR, denR, E);
  // ---- 7. combined edge+node attention accumulation -> S, s0 ----
  mis_edge_accum<<<512, 256, 0, stream>>>(f_src, f_dst, f_y, f_w, elogF, mxF, denF,
                                          nlogF, nmaxF, nsumF, drugs, SF, E);
  mis_edge_accum<<<512, 256, 0, stream>>>(r_src, r_dst, r_y, r_w, elogR, mxR, denR,
                                          nlogR, nmaxR, nsumR, drugs, SR, E);
  // ---- 8. attn-weighted feats sum -> F (for delta) ----
  mis_node_wsum<<<512, 256, 0, stream>>>(featF, nlogF, nmaxF, nsumF, FF, N);
  mis_node_wsum<<<512, 256, 0, stream>>>(featR, nlogR, nmaxR, nsumR, FR, N);
  // ---- 9. mixer / v_prior / delta_mean / z_refined ----
  mis_finalize<<<1, 512, 0, stream>>>(SF, SR, v_w, v_b, trust,
                                      mw1, mb1, mxa, mw2, mb2,
                                      zg, FF, FR,
                                      iw1, ib1, iap, iw2, ib2,
                                      nw, nb, dgate, out);
}
